// CustomGRULayer_18245021073416
// MI455X (gfx1250) — matrix operand order repaired and checked by reading, not compiled, not run on hardware
//
#include <hip/hip_runtime.h>
#include <hip/hip_bf16.h>
#include <math.h>

// ---------------------------------------------------------------------------
// BN-GRU for MI455X (gfx1250): bf16 WMMA GEMMs, fp32 accumulation.
// T=512 B=64 I=H=512.
//  - BN only touches input projections -> hoisted out of the recurrence and
//    fused with the W_* biases into one additive term per (t,b,col).
//  - Phase-1 GEMM is software-pipelined (ping-pong fragment buffers) so WMMA
//    overlaps the global loads instead of draining loadcnt each chunk.
//  - Recurrence runs as ONE persistent kernel: per-wave weight strips live in
//    registers (16x512 bf16 = 128 VGPRs; CDNA5 waves address 1024 VGPRs),
//    steps are separated by a device-wide atomic barrier instead of 1024
//    kernel launches.
// ---------------------------------------------------------------------------

typedef __attribute__((ext_vector_type(16))) __bf16 v16bf;
typedef __attribute__((ext_vector_type(8)))  __bf16 v8bf;
typedef __attribute__((ext_vector_type(8)))  float  v8f;

#define TT 512
#define BB 64
#define HH 512
#define KK 512
#define MM (TT*BB)      // 32768
#define NCAT 1536       // z | r | h sections
#define NBLK_REC 48     // 32 zr-role blocks + 16 h-role blocks (8 waves each)

// A fragment, 16x32 bf16, row-major source [*, ld].
// ISA 7.12.2: lanes 0-15 rows 0-15; VGPR0-3 = K 8*half..8*half+7,
//             VGPR4-7 = K 16+8*half .. 16+8*half+7 (2 bf16 per VGPR).
__device__ __forceinline__ v16bf load_fragA(const __bf16* p, int ld, int lane) {
  int row = lane & 15, half = lane >> 4;
  const __bf16* q = p + (size_t)row * ld + half * 8;
  v8bf lo = *(const v8bf*)(q);
  v8bf hi = *(const v8bf*)(q + 16);
  v16bf f;
#pragma unroll
  for (int i = 0; i < 8; ++i) { f[i] = lo[i]; f[i + 8] = hi[i]; }
  return f;
}

// B fragment, 32x16 bf16. Source is weight-style row-major [N, K] so
// B[k][n] = W[n][k]. Same K order as A: elements 0-7 = K 8*half..8*half+7,
// elements 8-15 = K 16+8*half..16+8*half+7, col = lane&15.
__device__ __forceinline__ v16bf load_fragB(const __bf16* p, int ld, int lane) {
  int col = lane & 15, half = lane >> 4;
  const __bf16* q = p + (size_t)col * ld + half * 8;
  v8bf lo = *(const v8bf*)(q);
  v8bf hi = *(const v8bf*)(q + 16);
  v16bf f;
#pragma unroll
  for (int i = 0; i < 8; ++i) { f[i] = lo[i]; f[i + 8] = hi[i]; }
  return f;
}

__device__ __forceinline__ v8f wmma_bf16(v16bf a, v16bf b, v8f c) {
  return __builtin_amdgcn_wmma_f32_16x16x32_bf16(false, a, false, b,
                                                 (short)0, c, false, false);
}

__device__ __forceinline__ float sigmoidf_(float x) {
  return 1.0f / (1.0f + __expf(-x));
}

// Device-wide barrier: monotonic counter, release/acquire via threadfence.
// Safe: NBLK_REC=48 blocks are always co-resident on MI455X.
__device__ __forceinline__ void gbar(int* cnt, int target) {
  __threadfence();                       // release this thread's stores
  __syncthreads();
  if (threadIdx.x == 0) {
    __hip_atomic_fetch_add(cnt, 1, __ATOMIC_RELEASE, __HIP_MEMORY_SCOPE_AGENT);
    while (__hip_atomic_load(cnt, __ATOMIC_ACQUIRE, __HIP_MEMORY_SCOPE_AGENT) < target)
      __builtin_amdgcn_s_sleep(1);
  }
  __syncthreads();
  __threadfence();                       // acquire: invalidate stale lines
}

// --------------------------- prep kernels ----------------------------------

__global__ void cvt_bf16_k(__bf16* __restrict__ dst, const float* __restrict__ src, int n) {
  int i = blockIdx.x * blockDim.x + threadIdx.x;
  if (i < n) dst[i] = (__bf16)src[i];
}

__global__ void init_state_k(const float* __restrict__ h0,
                             float* __restrict__ h_prev, __bf16* __restrict__ h_bf,
                             const float* __restrict__ uzb, const float* __restrict__ urb,
                             const float* __restrict__ uhb, float* __restrict__ ub_cat,
                             int* __restrict__ bar_cnt) {
  int i = blockIdx.x * blockDim.x + threadIdx.x;
  if (i < BB * HH) { float v = h0[i]; h_prev[i] = v; h_bf[i] = (__bf16)v; }
  if (i < HH) {
    ub_cat[i]        = uzb[i];
    ub_cat[HH + i]   = urb[i];
    ub_cat[2*HH + i] = uhb[i];
  }
  if (i == 0) *bar_cnt = 0;   // re-arm the device barrier every call
}

// --------------------------- phase 1: input GEMM ---------------------------
// u_cat[M,1536] = x_bf[M,512] @ Ucat_bf[1536,512]^T + ub_cat
// One wave owns a 16(M) x 64(N) tile. Two-deep ping-pong pipeline: while the
// WMMAs for chunk k run, the loads for chunk k+32 are already in flight.
__global__ void gemm_u_k(const __bf16* __restrict__ A, const __bf16* __restrict__ Bw,
                         const float* __restrict__ bias, float* __restrict__ Out) {
  int wid  = (blockIdx.x * blockDim.x + threadIdx.x) >> 5;
  int lane = threadIdx.x & 31;
  const int NT = NCAT / 64;               // 24
  int mt = wid / NT, nt = wid % NT;
  int m0 = mt * 16, n0 = nt * 64;
  v8f acc[4] = {};
  const __bf16* Ab = A  + (size_t)m0 * KK;
  const __bf16* Bb = Bw + (size_t)n0 * KK;

  v16bf aP, bP[4], aQ, bQ[4];             // ping / pong fragment sets
  aP = load_fragA(Ab, KK, lane);
#pragma unroll
  for (int j = 0; j < 4; ++j) bP[j] = load_fragB(Bb + (size_t)(j * 16) * KK, KK, lane);

#pragma unroll 1
  for (int k = 0; k < KK - 64; k += 64) {
    aQ = load_fragA(Ab + k + 32, KK, lane);
#pragma unroll
    for (int j = 0; j < 4; ++j)
      bQ[j] = load_fragB(Bb + (size_t)(j * 16) * KK + k + 32, KK, lane);
#pragma unroll
    for (int j = 0; j < 4; ++j) acc[j] = wmma_bf16(aP, bP[j], acc[j]);

    aP = load_fragA(Ab + k + 64, KK, lane);
#pragma unroll
    for (int j = 0; j < 4; ++j)
      bP[j] = load_fragB(Bb + (size_t)(j * 16) * KK + k + 64, KK, lane);
#pragma unroll
    for (int j = 0; j < 4; ++j) acc[j] = wmma_bf16(aQ, bQ[j], acc[j]);
  }
  // epilogue: chunk KK-64 sits in the P set; load final chunk KK-32 into Q
  aQ = load_fragA(Ab + KK - 32, KK, lane);
#pragma unroll
  for (int j = 0; j < 4; ++j)
    bQ[j] = load_fragB(Bb + (size_t)(j * 16) * KK + KK - 32, KK, lane);
#pragma unroll
  for (int j = 0; j < 4; ++j) acc[j] = wmma_bf16(aP, bP[j], acc[j]);
#pragma unroll
  for (int j = 0; j < 4; ++j) acc[j] = wmma_bf16(aQ, bQ[j], acc[j]);

  int nc = lane & 15, half = lane >> 4;
#pragma unroll
  for (int j = 0; j < 4; ++j) {
    int n = n0 + j * 16 + nc;
    float bv = bias[n];
    float* op = Out + (size_t)(m0 + half * 8) * NCAT + n;
#pragma unroll
    for (int i = 0; i < 8; ++i) op[(size_t)i * NCAT] = acc[j][i] + bv;
  }
}

// --------------------------- phase 2: BN (in place) ------------------------
// Per (t, col): mean/biased-var over batch (64 rows), rewrite as
// gamma*(u-mu)*rsqrt(var+eps) + beta + W_*_b  (full additive step term).
__global__ void bn_k(float* __restrict__ U,
                     const float* __restrict__ gamma, const float* __restrict__ beta,
                     const float* __restrict__ wzb, const float* __restrict__ wrb,
                     const float* __restrict__ whb) {
  int idx = blockIdx.x * blockDim.x + threadIdx.x;   // over T*1536
  if (idx >= TT * NCAT) return;
  int col = idx % NCAT, t = idx / NCAT;
  int sec = col >> 9, h = col & (HH - 1);
  float wb = (sec == 0) ? wzb[h] : (sec == 1) ? wrb[h] : whb[h];
  float g = gamma[h], be = beta[h];
  float* base = U + (size_t)t * BB * NCAT + col;
  float s = 0.f, sq = 0.f;
  for (int b = 0; b < BB; ++b) {
    float v = base[(size_t)b * NCAT];
    s += v; sq += v * v;
  }
  float mu  = s * (1.0f / BB);
  float var = sq * (1.0f / BB) - mu * mu;
  float inv = rsqrtf(var + 1e-5f) * g;
  for (int b = 0; b < BB; ++b) {
    float v = base[(size_t)b * NCAT];
    base[(size_t)b * NCAT] = (v - mu) * inv + be + wb;
  }
}

// --------------------- phase 3: persistent recurrence ----------------------
// Blocks 0..31 (256 waves): z|r GEMM, N=1024 -> tiles 4(M) x 64(N).
// Blocks 32..47 (128 waves): candidate GEMM, N=512 -> tiles 4(M) x 32(N).
// Each wave preloads its fixed 16x512 bf16 weight strip into 128 VGPRs ONCE;
// per step it only streams A fragments (h_bf / rh_bf, L2-resident): all 16
// loads issue as one clause, then a 16-deep WMMA chain retires them with
// progressively decreasing waits. Steps separated by device barriers.
__global__ void gru_rec_k(const __bf16* __restrict__ Wzr, const __bf16* __restrict__ Wh,
                          const float* __restrict__ Ucat,
                          float* __restrict__ h_prev, __bf16* __restrict__ h_bf,
                          float* __restrict__ z_buf, __bf16* __restrict__ rh_bf,
                          float* __restrict__ out, int* __restrict__ bar_cnt) {
  const int lane    = threadIdx.x & 31;
  const bool zr_role = blockIdx.x < 32;
  const int wid = ((zr_role ? blockIdx.x : blockIdx.x - 32) << 3) + (threadIdx.x >> 5);
  const int m0 = (wid & 3) * 16;
  const int n0 = (wid >> 2) * 16;        // zr: 0..63 tiles, h: 0..31 tiles

  // register-resident weight strip (constant across all 512 steps)
  const __bf16* Bbase = (zr_role ? Wzr : Wh) + (size_t)n0 * KK;
  v16bf bw[16];
#pragma unroll
  for (int kc = 0; kc < 16; ++kc) bw[kc] = load_fragB(Bbase + kc * 32, KK, lane);

  const int nc = n0 + (lane & 15);
  const int mb = m0 + (lane >> 4) * 8;
  int nbar = 0;

  for (int t = 0; t < TT; ++t) {
    // ---- phase A: z = sigmoid(h@Wz^T + Az), r = sigmoid(h@Wr^T + Ar) ----
    if (zr_role) {
      v16bf a[16];
#pragma unroll
      for (int kc = 0; kc < 16; ++kc)
        a[kc] = load_fragA(h_bf + (size_t)m0 * KK + kc * 32, KK, lane);
      v8f acc = {};
#pragma unroll
      for (int kc = 0; kc < 16; ++kc) acc = wmma_bf16(a[kc], bw[kc], acc);
      const float* arow = Ucat + (size_t)t * BB * NCAT;
#pragma unroll
      for (int i = 0; i < 8; ++i) {
        int m = mb + i;
        float v = sigmoidf_(acc[i] + arow[(size_t)m * NCAT + nc]);
        if (nc < HH) z_buf[m * HH + nc] = v;
        else {
          int hc = nc - HH;
          rh_bf[m * HH + hc] = (__bf16)(v * h_prev[m * HH + hc]);
        }
      }
    }
    ++nbar; gbar(bar_cnt, nbar * NBLK_REC);

    // ---- phase B: nh = tanh(rh@Wh^T + Ah); h = (1-z)h + z*nh ----
    if (!zr_role) {
      v16bf a[16];
#pragma unroll
      for (int kc = 0; kc < 16; ++kc)
        a[kc] = load_fragA(rh_bf + (size_t)m0 * KK + kc * 32, KK, lane);
      v8f acc = {};
#pragma unroll
      for (int kc = 0; kc < 16; ++kc) acc = wmma_bf16(a[kc], bw[kc], acc);
      const float* arow = Ucat + (size_t)t * BB * NCAT + 2 * HH;
#pragma unroll
      for (int i = 0; i < 8; ++i) {
        int m = mb + i;
        float nh = tanhf(acc[i] + arow[(size_t)m * NCAT + nc]);
        float z  = z_buf[m * HH + nc];
        float hp = h_prev[m * HH + nc];
        float hn = (1.0f - z) * hp + z * nh;
        h_prev[m * HH + nc] = hn;
        h_bf[m * HH + nc]   = (__bf16)hn;
        size_t o = (size_t)t * BB * HH + (size_t)m * HH + nc;
        out[o] = hn;                                  // output [T,B,1,H]
        out[(size_t)TT * BB * HH + o] = hn;           // h_out  [1,T,B,H]
      }
    }
    ++nbar; gbar(bar_cnt, nbar * NBLK_REC);
  }
}

// ---------------------------------------------------------------------------

extern "C" void kernel_launch(void* const* d_in, const int* in_sizes, int n_in,
                              void* d_out, int out_size, void* d_ws, size_t ws_size,
                              hipStream_t stream) {
  (void)in_sizes; (void)n_in; (void)out_size; (void)ws_size;
  const float* x    = (const float*)d_in[0];
  const float* h0   = (const float*)d_in[1];
  const float* Wz   = (const float*)d_in[2];
  const float* Wzb  = (const float*)d_in[3];
  const float* Uz   = (const float*)d_in[4];
  const float* Uzb  = (const float*)d_in[5];
  const float* Wr   = (const float*)d_in[6];
  const float* Wrb  = (const float*)d_in[7];
  const float* Ur   = (const float*)d_in[8];
  const float* Urb  = (const float*)d_in[9];
  const float* Wh   = (const float*)d_in[10];
  const float* Whb  = (const float*)d_in[11];
  const float* Uh   = (const float*)d_in[12];
  const float* Uhb  = (const float*)d_in[13];
  const float* gam  = (const float*)d_in[14];
  const float* bet  = (const float*)d_in[15];
  float* out = (float*)d_out;

  // workspace carve-up (256B aligned)
  char* ws = (char*)d_ws;
  size_t off = 0;
  auto carve = [&](size_t bytes) -> void* {
    void* p = ws + off; off += (bytes + 255) & ~(size_t)255; return p;
  };
  float*  u_cat   = (float*) carve((size_t)MM * NCAT * 4);   // 192 MB
  __bf16* x_bf    = (__bf16*)carve((size_t)MM * KK * 2);     // 32 MB
  __bf16* Ucat_bf = (__bf16*)carve((size_t)NCAT * KK * 2);   // 1.5 MB
  __bf16* Wzr_bf  = (__bf16*)carve((size_t)2 * HH * KK * 2); // 1 MB
  __bf16* Wh_bf   = (__bf16*)carve((size_t)HH * KK * 2);     // 0.5 MB
  float*  ub_cat  = (float*) carve((size_t)NCAT * 4);
  float*  h_prev  = (float*) carve((size_t)BB * HH * 4);
  __bf16* h_bf    = (__bf16*)carve((size_t)BB * HH * 2);
  float*  z_buf   = (float*) carve((size_t)BB * HH * 4);
  __bf16* rh_bf   = (__bf16*)carve((size_t)BB * HH * 2);
  int*    bar_cnt = (int*)   carve(256);

  const int W = HH * KK;                   // 262144 elems per weight matrix
  // bf16 conversions
  cvt_bf16_k<<<(MM * KK + 255) / 256, 256, 0, stream>>>(x_bf, x, MM * KK);
  cvt_bf16_k<<<(W + 255) / 256, 256, 0, stream>>>(Ucat_bf,          Uz, W);
  cvt_bf16_k<<<(W + 255) / 256, 256, 0, stream>>>(Ucat_bf + W,      Ur, W);
  cvt_bf16_k<<<(W + 255) / 256, 256, 0, stream>>>(Ucat_bf + 2 * W,  Uh, W);
  cvt_bf16_k<<<(W + 255) / 256, 256, 0, stream>>>(Wzr_bf,           Wz, W);
  cvt_bf16_k<<<(W + 255) / 256, 256, 0, stream>>>(Wzr_bf + W,       Wr, W);
  cvt_bf16_k<<<(W + 255) / 256, 256, 0, stream>>>(Wh_bf,            Wh, W);
  init_state_k<<<(BB * HH + 255) / 256, 256, 0, stream>>>(h0, h_prev, h_bf,
                                                          Uzb, Urb, Uhb, ub_cat,
                                                          bar_cnt);

  // phase 1: u_cat = x @ Ucat^T + b   (49152 waves, 16x64 tiles, pipelined)
  gemm_u_k<<<(MM / 16) * (NCAT / 64) / 8, 256, 0, stream>>>(x_bf, Ucat_bf, ub_cat, u_cat);

  // phase 2: in-place BN + W-bias fold
  bn_k<<<(TT * NCAT + 255) / 256, 256, 0, stream>>>(u_cat, gam, bet, Wzb, Wrb, Whb);

  // phase 3: one persistent kernel runs all 512 steps with device barriers
  gru_rec_k<<<NBLK_REC, 256, 0, stream>>>(Wzr_bf, Wh_bf, u_cat, h_prev, h_bf,
                                          z_buf, rh_bf, out, bar_cnt);
}
